// EngramBranchSample_37812892074116
// MI455X (gfx1250) — compile-verified, matrix operand order checked by reading, not run on hardware
//
#include <hip/hip_runtime.h>
#include <hip/hip_bf16.h>

// ---------------- problem constants ----------------
#define B_   4
#define T_   4096
#define D_   2048
#define DB_  512
#define BT_  (B_*T_)          // 16384 tokens
#define KC_  4                // conv taps

typedef __bf16 bf16;
typedef __attribute__((ext_vector_type(16))) __bf16 v16bf;
typedef __attribute__((ext_vector_type(8)))  float  v8f;

union FragU { v16bf v; uint4 q[2]; };

// ---------------- bf16 GEMM: C[M,N] = A[M,K] * Bm[N,K]^T ----------------
#define TILE_M 128
#define TILE_N 128
#define TILE_K 32
#define LDS_STRIDE 40   // 32 data halves + 8 pad -> conflict-free 16B LDS reads

// CDNA5 async copy: global memory to LDS, ASYNCcnt-tracked.
// Each call copies this lane's 32 bytes (two b128 chunks; ioffset applies to both sides).
__device__ __forceinline__ void async_copy_32B(unsigned lds_off, const void* gaddr) {
  asm volatile("global_load_async_to_lds_b128 %0, %1, off\n\t"
               "global_load_async_to_lds_b128 %0, %1, off offset:16"
               :: "v"(lds_off), "v"(gaddr) : "memory");
}
__device__ __forceinline__ void wait_async0() {
  asm volatile("s_wait_asynccnt 0x0" ::: "memory");
}

__device__ __forceinline__ v16bf lds_frag(const unsigned short* lds, int rcBase, int lane) {
  // CDNA5 16-bit A/B operand layout:
  //   lanes 0-15 : row rcBase+lane,    elems 0..7 = K 0..7,  elems 8..15 = K 16..23
  //   lanes 16-31: row rcBase+lane-16, elems 0..7 = K 8..15, elems 8..15 = K 24..31
  const int r   = rcBase + (lane & 15);
  const int klo = (lane >> 4) << 3;
  const unsigned short* p = lds + r * LDS_STRIDE + klo;
  FragU u;
  u.q[0] = *(const uint4*)(p);        // K klo .. klo+7
  u.q[1] = *(const uint4*)(p + 16);   // K klo+16 .. klo+23
  return u.v;
}

template <bool OUT_BF16>
__global__ __launch_bounds__(256)
void gemm_bf16_kernel(const bf16* __restrict__ A, const bf16* __restrict__ Bm,
                      void* __restrict__ Cout, int M, int N, int K) {
  // double-buffered tiles: 2 * (128*40 + 128*40) halves = 40 KB LDS
  __shared__ __align__(16) unsigned short ldsA[2][TILE_M * LDS_STRIDE];
  __shared__ __align__(16) unsigned short ldsB[2][TILE_N * LDS_STRIDE];

  const int tid   = threadIdx.x;
  const int lane  = tid & 31;
  const int wave  = tid >> 5;
  const int waveM = wave & 3;     // 4 waves along M  -> 32 rows each
  const int waveN = wave >> 2;    // 2 waves along N  -> 64 cols each
  const int blockRow = blockIdx.y * TILE_M;
  const int blockCol = blockIdx.x * TILE_N;

  // tile loader assignment: 256 threads = 128 rows x 2 16-half segments
  const int ldRow = tid >> 1;
  const int ldSeg = (tid & 1) * 16;        // half-element offset within row

  const unsigned ldsAbase = (unsigned)(size_t)&ldsA[0][ldRow * LDS_STRIDE + ldSeg];
  const unsigned ldsBbase = (unsigned)(size_t)&ldsB[0][ldRow * LDS_STRIDE + ldSeg];
  const unsigned bufStep  = (unsigned)(TILE_M * LDS_STRIDE * sizeof(unsigned short));

  const bf16* gA = A  + (size_t)(blockRow + ldRow) * K + ldSeg;
  const bf16* gB = Bm + (size_t)(blockCol + ldRow) * K + ldSeg;

  auto issue_tile = [&](int buf, int k0) {
    async_copy_32B(ldsAbase + (unsigned)buf * bufStep, gA + k0);
    async_copy_32B(ldsBbase + (unsigned)buf * bufStep, gB + k0);
  };

  v8f acc[2][4];
  v8f zero8 = {0.f,0.f,0.f,0.f,0.f,0.f,0.f,0.f};
  #pragma unroll
  for (int i = 0; i < 2; ++i)
    #pragma unroll
    for (int j = 0; j < 4; ++j) acc[i][j] = zero8;

  const int kTiles = K / TILE_K;

  // prologue: async-stage tile 0
  issue_tile(0, 0);
  wait_async0();
  __syncthreads();

  for (int kt = 0; kt < kTiles; ++kt) {
    const int cur = kt & 1;
    if (kt + 1 < kTiles) {
      issue_tile(cur ^ 1, (kt + 1) * TILE_K);          // overlap next tile with WMMA
      if (kt + 2 < kTiles) {                           // warm GL2 two tiles ahead
        __builtin_prefetch(gA + (kt + 2) * TILE_K, 0, 1);
        __builtin_prefetch(gB + (kt + 2) * TILE_K, 0, 1);
      }
    }

    v16bf af[2], bf_[4];
    #pragma unroll
    for (int i = 0; i < 2; ++i) af[i]  = lds_frag(&ldsA[cur][0], waveM * 32 + i * 16, lane);
    #pragma unroll
    for (int j = 0; j < 4; ++j) bf_[j] = lds_frag(&ldsB[cur][0], waveN * 64 + j * 16, lane);

    #pragma unroll
    for (int i = 0; i < 2; ++i)
      #pragma unroll
      for (int j = 0; j < 4; ++j)
        acc[i][j] = __builtin_amdgcn_wmma_f32_16x16x32_bf16(
            false, af[i], false, bf_[j], (short)0, acc[i][j], false, false);

    wait_async0();       // our async writes into the next buffer are done
    __syncthreads();     // everyone's are done -> safe to read next / overwrite cur
  }

  // epilogue: 16x16 f32 D layout -> lane holds col=lane&15, rows e + (lane>>4)*8
  const int mBase  = blockRow + waveM * 32;
  const int nBase  = blockCol + waveN * 64;
  const int col    = lane & 15;
  const int rowOff = (lane >> 4) * 8;
  #pragma unroll
  for (int i = 0; i < 2; ++i)
    #pragma unroll
    for (int j = 0; j < 4; ++j)
      #pragma unroll
      for (int e = 0; e < 8; ++e) {
        const size_t m = (size_t)(mBase + i * 16 + rowOff + e);
        const size_t n = (size_t)(nBase + j * 16 + col);
        const float val = acc[i][j][e];
        if (OUT_BF16) ((bf16*)Cout)[m * N + n] = (bf16)val;
        else          ((float*)Cout)[m * N + n] = val;
      }
}

// ---------------- elementwise kernels ----------------
__global__ void cvt_f32_bf16_kernel(const float* __restrict__ src,
                                    bf16* __restrict__ dst, int n) {
  const int i = blockIdx.x * blockDim.x + threadIdx.x;
  if (i < n) dst[i] = (bf16)src[i];
}

// W_mix [3, DB, DB] (i,f,e) -> Wcat bf16 [DB, 3*DB] with layout [f][i*DB+e]
__global__ void pack_wmix_kernel(const float* __restrict__ src, bf16* __restrict__ dst) {
  const int gid = blockIdx.x * blockDim.x + threadIdx.x;
  if (gid >= 3 * DB_ * DB_) return;
  const int e = gid & (DB_ - 1);
  const int f = (gid >> 9) & (DB_ - 1);
  const int i = gid >> 18;
  dst[(size_t)f * (3 * DB_) + i * DB_ + e] = (bf16)src[gid];
}

// U[bt, i*DB+e] = (sum of doc-masked shifts of z up to order_i) / (order_i * 3), bf16
__global__ void build_u_kernel(const float* __restrict__ z, const int* __restrict__ doc,
                               bf16* __restrict__ U) {
  const int gid = blockIdx.x * blockDim.x + threadIdx.x;
  if (gid >= BT_ * DB_) return;
  const int e  = gid & (DB_ - 1);
  const int bt = gid >> 9;
  const int t  = bt & (T_ - 1);
  const int d0 = doc[bt];
  const float z0 = z[(size_t)bt * DB_ + e];
  float z1 = 0.f, z2 = 0.f, z3 = 0.f;
  if (t >= 1 && doc[bt - 1] == d0) z1 = z[(size_t)(bt - 1) * DB_ + e];
  if (t >= 2 && doc[bt - 2] == d0) z2 = z[(size_t)(bt - 2) * DB_ + e];
  if (t >= 3 && doc[bt - 3] == d0) z3 = z[(size_t)(bt - 3) * DB_ + e];
  const float t2 = z0 + z1;
  const float t3 = t2 + z2;
  const float t4 = t3 + z3;
  const size_t base = (size_t)bt * (3 * DB_);
  U[base + e]            = (bf16)(t2 * (1.f / 6.f));   // order 2, /3
  U[base + DB_ + e]      = (bf16)(t3 * (1.f / 9.f));   // order 3, /3
  U[base + 2 * DB_ + e]  = (bf16)(t4 * (1.f / 12.f));  // order 4, /3
}

// per-token: gate = sigmoid(signed-sqrt( <rms(x),rms(k)>/sqrt(D) ));
// out = gate*v; normed = rms(out) = g*v * rsqrt(g^2*mean(v^2)+eps)  (bf16 out)
__global__ __launch_bounds__(256)
void gate_norm_kernel(const float* __restrict__ x, const bf16* __restrict__ kh,
                      const bf16* __restrict__ vh, bf16* __restrict__ nh) {
  const int bt  = blockIdx.x;
  const int tid = threadIdx.x;
  const size_t base = (size_t)bt * D_;
  float sxk = 0.f, sxx = 0.f, skk = 0.f, svv = 0.f;
  for (int d = tid; d < D_; d += 256) {
    const float xv = x[base + d];
    const float kv = (float)kh[base + d];
    const float vv = (float)vh[base + d];
    sxk += xv * kv; sxx += xv * xv; skk += kv * kv; svv += vv * vv;
  }
  __shared__ float r0[256], r1[256], r2[256], r3[256];
  r0[tid] = sxk; r1[tid] = sxx; r2[tid] = skk; r3[tid] = svv;
  __syncthreads();
  for (int s = 128; s > 0; s >>= 1) {
    if (tid < s) {
      r0[tid] += r0[tid + s]; r1[tid] += r1[tid + s];
      r2[tid] += r2[tid + s]; r3[tid] += r3[tid + s];
    }
    __syncthreads();
  }
  __shared__ float sG, sRn;
  if (tid == 0) {
    const float Df = (float)D_;
    float gl = r0[0] * rsqrtf(r1[0] / Df + 1e-6f) * rsqrtf(r2[0] / Df + 1e-6f)
               / sqrtf(Df);
    const float sgn = (gl > 0.f) ? 1.f : ((gl < 0.f) ? -1.f : 0.f);
    gl = sgn * sqrtf(fmaxf(fabsf(gl), 1e-6f));
    const float g = 1.f / (1.f + __expf(-gl));
    sG  = g;
    sRn = rsqrtf(g * g * r3[0] / Df + 1e-6f);
  }
  __syncthreads();
  const float g = sG, rn = sRn;
  for (int d = tid; d < D_; d += 256) {
    const float vv = (float)vh[base + d];
    nh[base + d] = (bf16)(g * vv * rn);
  }
}

// depthwise causal doc-masked conv (K=4 taps) + SiLU, f32 out
__global__ void conv_silu_kernel(const bf16* __restrict__ nh, const int* __restrict__ doc,
                                 const float* __restrict__ cw, float* __restrict__ out) {
  const int gid = blockIdx.x * blockDim.x + threadIdx.x;
  if (gid >= BT_ * D_) return;
  const int d  = gid & (D_ - 1);
  const int bt = gid >> 11;
  const int t  = bt & (T_ - 1);
  const int d0 = doc[bt];
  float res = (float)nh[(size_t)bt * D_ + d] * cw[d * KC_ + 3];
  if (t >= 1 && doc[bt - 1] == d0) res += (float)nh[(size_t)(bt - 1) * D_ + d] * cw[d * KC_ + 2];
  if (t >= 2 && doc[bt - 2] == d0) res += (float)nh[(size_t)(bt - 2) * D_ + d] * cw[d * KC_ + 1];
  if (t >= 3 && doc[bt - 3] == d0) res += (float)nh[(size_t)(bt - 3) * D_ + d] * cw[d * KC_ + 0];
  out[gid] = res / (1.f + __expf(-res));   // silu
}

// ---------------- launcher ----------------
extern "C" void kernel_launch(void* const* d_in, const int* in_sizes, int n_in,
                              void* d_out, int out_size, void* d_ws, size_t ws_size,
                              hipStream_t stream) {
  const float* x      = (const float*)d_in[0];   // [B,T,D]
  const int*   doc    = (const int*)  d_in[1];   // [B,T]
  const float* W_in   = (const float*)d_in[2];   // [DB,D]
  const float* W_mix  = (const float*)d_in[3];   // [3,DB,DB]
  const float* W_k    = (const float*)d_in[4];   // [D,DB]
  const float* W_v    = (const float*)d_in[5];   // [D,DB]
  const float* conv_w = (const float*)d_in[6];   // [D,K]
  float* out = (float*)d_out;                    // [B,T,D] f32

  char* p = (char*)d_ws;
  auto carve = [&](size_t bytes) -> char* {
    char* r = p;
    p += (bytes + 255) & ~(size_t)255;
    return r;
  };
  bf16*  xh  = (bf16*) carve((size_t)BT_ * D_ * 2);        // x in bf16
  bf16*  wih = (bf16*) carve((size_t)DB_ * D_ * 2);        // W_in bf16
  bf16*  wmh = (bf16*) carve((size_t)DB_ * 3 * DB_ * 2);   // W_mix concat bf16
  bf16*  wkh = (bf16*) carve((size_t)D_ * DB_ * 2);        // W_k bf16
  bf16*  wvh = (bf16*) carve((size_t)D_ * DB_ * 2);        // W_v bf16
  float* z   = (float*)carve((size_t)BT_ * DB_ * 4);       // bottleneck f32
  bf16*  U   = (bf16*) carve((size_t)BT_ * 3 * DB_ * 2);   // shift-avg concat bf16
  float* y   = (float*)carve((size_t)BT_ * DB_ * 4);       // mixed f32
  bf16*  yh  = (bf16*) carve((size_t)BT_ * DB_ * 2);       // y bf16
  bf16*  kh  = (bf16*) carve((size_t)BT_ * D_ * 2);        // k bf16
  bf16*  vh  = (bf16*) carve((size_t)BT_ * D_ * 2);        // v bf16
  bf16*  nh  = (bf16*) carve((size_t)BT_ * D_ * 2);        // rms(out) bf16
  (void)ws_size; (void)in_sizes; (void)n_in; (void)out_size;

  const int threads = 256;
  auto blocks = [](size_t n, int t) { return (unsigned)((n + t - 1) / t); };

  // 0) precision packing
  cvt_f32_bf16_kernel<<<blocks((size_t)BT_ * D_, threads), threads, 0, stream>>>(x, xh, BT_ * D_);
  cvt_f32_bf16_kernel<<<blocks((size_t)DB_ * D_, threads), threads, 0, stream>>>(W_in, wih, DB_ * D_);
  cvt_f32_bf16_kernel<<<blocks((size_t)D_ * DB_, threads), threads, 0, stream>>>(W_k, wkh, D_ * DB_);
  cvt_f32_bf16_kernel<<<blocks((size_t)D_ * DB_, threads), threads, 0, stream>>>(W_v, wvh, D_ * DB_);
  pack_wmix_kernel<<<blocks((size_t)3 * DB_ * DB_, threads), threads, 0, stream>>>(W_mix, wmh);

  // 1) z = x * W_in^T   [16384,2048] x [512,2048]^T
  {
    dim3 grid(DB_ / TILE_N, BT_ / TILE_M);
    gemm_bf16_kernel<false><<<grid, threads, 0, stream>>>(xh, wih, z, BT_, DB_, D_);
  }

  // 2) doc-masked shift averages, concatenated along K
  build_u_kernel<<<blocks((size_t)BT_ * DB_, threads), threads, 0, stream>>>(z, doc, U);

  // 3) y = U * Wcat^T   [16384,1536] x [512,1536]^T
  {
    dim3 grid(DB_ / TILE_N, BT_ / TILE_M);
    gemm_bf16_kernel<false><<<grid, threads, 0, stream>>>(U, wmh, y, BT_, DB_, 3 * DB_);
  }
  cvt_f32_bf16_kernel<<<blocks((size_t)BT_ * DB_, threads), threads, 0, stream>>>(y, yh, BT_ * DB_);

  // 4) k = y * W_k^T, v = y * W_v^T   [16384,512] x [2048,512]^T  (bf16 out)
  {
    dim3 grid(D_ / TILE_N, BT_ / TILE_M);
    gemm_bf16_kernel<true><<<grid, threads, 0, stream>>>(yh, wkh, kh, BT_, D_, DB_);
    gemm_bf16_kernel<true><<<grid, threads, 0, stream>>>(yh, wvh, vh, BT_, D_, DB_);
  }

  // 5) gate + RMS-normalized gated output (one block per token)
  gate_norm_kernel<<<BT_, threads, 0, stream>>>(x, kh, vh, nh);

  // 6) depthwise causal conv + SiLU -> f32 output
  conv_silu_kernel<<<blocks((size_t)BT_ * D_, threads), threads, 0, stream>>>(nh, doc, conv_w, out);
}
